// SymbolicEncoder_128849019169
// MI455X (gfx1250) — matrix operand order repaired and checked by reading, not compiled, not run on hardware
//
#include <hip/hip_runtime.h>
#include <cstdint>
#include <cstddef>

// ---------- CDNA5 WMMA plumbing (wave32, 16x16x32 bf16 -> f32) ----------
typedef __attribute__((ext_vector_type(16))) __bf16 bf16x16;
typedef __attribute__((ext_vector_type(8)))  float  f32x8;
typedef __attribute__((ext_vector_type(4)))  unsigned int u32x4;
typedef __attribute__((ext_vector_type(8)))  int i32x8;
typedef __attribute__((ext_vector_type(4)))  int i32x4;

union FragU { bf16x16 v; u32x4 q[2]; };

__device__ __forceinline__ f32x8 wmma_bf16(bf16x16 a, bf16x16 b, f32x8 c) {
  return __builtin_amdgcn_wmma_f32_16x16x32_bf16(false, a, false, b, (short)0, c,
                                                 false, false);
}
__device__ __forceinline__ __bf16 f2bf(float f) { return (__bf16)f; }

// ---------- Tensor Data Mover: 1-row (contiguous) tile load into LDS ----------
#define TDM_HAVE __has_builtin(__builtin_amdgcn_tensor_load_to_lds)

#if TDM_HAVE
// D# layout per CDNA5 ISA (normal mode, 1 row tile):
//  group0: [1:0]=count=1, [63:32]=lds_addr, [120:64]=global_addr, [127:126]=type=2
//  group1: data_size=1 (2B elems); tensor_dim0=nelem; tensor_dim1=1;
//          tile_dim0=nelem; tile_dim1/2=0; stride0=nelem
// clang-23 toolchain: 6-arg builtin (g0 u32x4, g1 i32x8, g2 i32x4, g3 i32x4,
//                                    extra i32x8, i32 cpol)
__device__ __forceinline__ void tdm_load_row(unsigned ldsOff, const void* g,
                                             unsigned nelem /* bf16 elems <= 65535 */) {
  const unsigned long long ga = (unsigned long long)g;
  u32x4 g0 = {0u, 0u, 0u, 0u};
  g0.x = 1u;                                       // count=1, user descriptor
  g0.y = ldsOff;                                   // LDS byte address
  g0.z = (unsigned)ga;                             // global_addr[31:0]
  g0.w = (unsigned)((ga >> 32) & 0x01ffffffu) | (2u << 30);  // addr[56:32] | type=2
  i32x8 g1 = {0, 0, 0, 0, 0, 0, 0, 0};
  g1[0] = (int)(1u << 16);                         // data_size = 1 -> 2 bytes
  g1[1] = (int)((nelem & 0xffffu) << 16);          // tensor_dim0[15:0]
  g1[2] = (int)(((nelem >> 16) & 0xffffu) | (1u << 16)); // dim0[31:16], tensor_dim1=1
  g1[3] = (int)((nelem & 0xffffu) << 16);          // tile_dim0 = nelem
  g1[4] = 0;                                       // tile_dim1 = tile_dim2 = 0 (1D)
  g1[5] = (int)nelem;                              // tensor_dim0_stride[31:0]
  const i32x4 gz4 = {0, 0, 0, 0};                  // groups 2/3 unused (<=2D)
  const i32x8 gz8 = {0, 0, 0, 0, 0, 0, 0, 0};
  __builtin_amdgcn_tensor_load_to_lds(g0, g1, gz4, gz4, gz8, 0);
}
#endif

// ---------- problem constants ----------
#define P1 524288L   // 32*128*128 pixels after conv1
#define P2 131072L   // 32*64*64
#define P3 32768L    // 32*32*32
#define DQ 64
#define KCODE 512

// workspace layout (bytes)
static const size_t OFF_RAW   = 0;                      // max P1*64*4 = 134217728
static const size_t OFF_ACT1  = 134217728;              // bf16 P1*64 (reused for h3)
static const size_t OFF_ACT2  = OFF_ACT1 + 67108864;    // bf16 P2*128
static const size_t OFF_ZF    = OFF_ACT2 + 33554432;    // fp32 P3*64
static const size_t OFF_ZB    = OFF_ZF   + 8388608;     // bf16 P3*64
static const size_t OFF_BT2   = OFF_ZB   + 4194304;     // bf16 128x1024
static const size_t OFF_BT3   = OFF_BT2  + 262144;      // bf16 256x2048
static const size_t OFF_BTQ   = OFF_BT3  + 1048576;     // bf16 64x256
static const size_t OFF_BTC   = OFF_BTQ  + 32768;       // bf16 512x64
static const size_t OFF_CNORM = OFF_BTC  + 65536;       // fp32 512
static const size_t OFF_STATS = OFF_CNORM + 2048;

// stats region (float indices)
#define S_SUM1 0
#define S_SQ1  64
#define S_SC1  128
#define S_SH1  192
#define S_SUM2 256
#define S_SQ2  384
#define S_SC2  512
#define S_SH2  640
#define S_SUM3 768
#define S_SQ3  1024
#define S_SC3  1280
#define S_SH3  1536
#define S_CNT  1792
#define S_LOSS 2304
#define S_TOTAL 2305

// ---------- tiny utility kernels ----------
__global__ void k_zero(float* __restrict__ p, int n) {
  int i = blockIdx.x * blockDim.x + threadIdx.x;
  if (i < n) p[i] = 0.0f;
}

__global__ void k_cast(const float* __restrict__ in, __bf16* __restrict__ out, int n) {
  int i = blockIdx.x * blockDim.x + threadIdx.x;
  if (i < n) out[i] = f2bf(in[i]);
}

// OIHW (COUT,CIN,4,4) fp32 -> BT[n=o][k=(r*4+s)*CIN+c] bf16
__global__ void k_wconv(const float* __restrict__ w, __bf16* __restrict__ bt,
                        int CIN, int COUT) {
  int i = blockIdx.x * blockDim.x + threadIdx.x;
  int total = COUT * CIN * 16;
  if (i >= total) return;
  int s = i & 3, r = (i >> 2) & 3;
  int c = (i >> 4) % CIN;
  int o = (i >> 4) / CIN;
  bt[(long)o * (CIN * 16) + (r * 4 + s) * CIN + c] = f2bf(w[i]);
}

__global__ void k_cnorm(const float* __restrict__ cb, float* __restrict__ cn) {
  int t = blockIdx.x * blockDim.x + threadIdx.x;
  if (t < KCODE) {
    float s = 0.0f;
    for (int d = 0; d < DQ; ++d) { float v = cb[t * DQ + d]; s += v * v; }
    cn[t] = s;
  }
}

// ---------- conv1: direct fp32 (Cin=3, K=48 too skinny for WMMA) ----------
__global__ void k_conv1(const float* __restrict__ x, const float* __restrict__ w,
                        const float* __restrict__ b, float* __restrict__ raw) {
  const long i = (long)blockIdx.x * blockDim.x + threadIdx.x;
  if (i >= P1 * 64) return;
  const int co = (int)(i & 63);
  const long P = i >> 6;
  const int bimg = (int)(P >> 14);
  const int rem  = (int)(P & 16383);
  const int oy = rem >> 7, ox = rem & 127;
  float acc = b[co];
  for (int c = 0; c < 3; ++c)
    for (int r = 0; r < 4; ++r) {
      const int iy = 2 * oy - 1 + r;
      if (iy < 0 || iy >= 256) continue;
      for (int s = 0; s < 4; ++s) {
        const int ix = 2 * ox - 1 + s;
        if (ix < 0 || ix >= 256) continue;
        acc += x[(((long)bimg * 3 + c) * 256 + iy) * 256 + ix] *
               w[((co * 3 + c) * 4 + r) * 4 + s];
      }
    }
  raw[i] = acc;  // NHWC
}

// ---------- batchnorm stats / finalize / apply+relu+bf16 ----------
__global__ void k_bnstats(const float* __restrict__ raw, float* __restrict__ sum,
                          float* __restrict__ sq, long total, int cmask) {
  const long i0 = (long)blockIdx.x * blockDim.x + threadIdx.x;
  const long stride = (long)gridDim.x * blockDim.x;   // multiple of C
  const int c = (int)(i0 & cmask);
  float s = 0.0f, s2 = 0.0f;
  for (long i = i0; i < total; i += stride) { float v = raw[i]; s += v; s2 += v * v; }
  atomicAdd(&sum[c], s);
  atomicAdd(&sq[c], s2);
}

__global__ void k_bnfin(const float* __restrict__ sum, const float* __restrict__ sq,
                        const float* __restrict__ g, const float* __restrict__ be,
                        float* __restrict__ scale, float* __restrict__ shift,
                        float cntInv, int C) {
  int c = blockIdx.x * blockDim.x + threadIdx.x;
  if (c >= C) return;
  float m = sum[c] * cntInv;
  float v = sq[c] * cntInv - m * m;
  float inv = rsqrtf(v + 1e-5f);
  float sc = g[c] * inv;
  scale[c] = sc;
  shift[c] = be[c] - m * sc;
}

__global__ void k_act(const float* __restrict__ raw, const float* __restrict__ scale,
                      const float* __restrict__ shift, __bf16* __restrict__ out,
                      long total, int cmask) {
  const long stride = (long)gridDim.x * blockDim.x;
  for (long i = (long)blockIdx.x * blockDim.x + threadIdx.x; i < total; i += stride) {
    const int c = (int)(i & cmask);
    float v = raw[i] * scale[c] + shift[c];
    out[i] = f2bf(v > 0.0f ? v : 0.0f);
  }
}

// ---------- WMMA 4x4 stride-2 conv: im2col GEMM, M=32 tile (B-frag reuse x2) ----------
template<int CIN, int COUT, int HO, int WO>
__global__ __launch_bounds__(128) void k_conv4x4s2_wmma(
    const __bf16* __restrict__ act,   // NHWC (B, 2HO, 2WO, CIN)
    const __bf16* __restrict__ bt,    // (COUT, 16*CIN)
    const float*  __restrict__ bias,
    float* __restrict__ out)          // NHWC (B*HO*WO, COUT) fp32
{
  constexpr int KK = 16 * CIN;
  constexpr int HI = 2 * HO, WI = 2 * WO;
  __shared__ __align__(16) __bf16 sA[32 * KK];   // 32-pixel im2col panel
  const int lane = threadIdx.x & 31;
  const int wave = threadIdx.x >> 5;
  const long pixBase = (long)blockIdx.x * 32;
  const u32x4 Z4 = {0u, 0u, 0u, 0u};

  for (int job = threadIdx.x; job < 512; job += 128) {   // 32 pixels x 16 taps
    const int p = job >> 4, t = job & 15;
    const int r = t >> 2, s = t & 3;
    const long P = pixBase + p;
    const int bimg = (int)(P / (HO * WO));
    const int rem  = (int)(P % (HO * WO));
    const int oy = rem / WO, ox = rem % WO;
    const int iy = 2 * oy - 1 + r, ix = 2 * ox - 1 + s;
    __bf16* dst = sA + p * KK + t * CIN;
    if (iy >= 0 && iy < HI && ix >= 0 && ix < WI) {
      const __bf16* src = act + (((long)bimg * HI + iy) * WI + ix) * CIN;
      #pragma unroll
      for (int c = 0; c < CIN; c += 8)
        *reinterpret_cast<u32x4*>(dst + c) = *reinterpret_cast<const u32x4*>(src + c);
    } else {
      #pragma unroll
      for (int c = 0; c < CIN; c += 8)
        *reinterpret_cast<u32x4*>(dst + c) = Z4;
    }
  }
  __syncthreads();

  const int hf = lane >> 4, nn = lane & 15;
  const int n0 = (blockIdx.y * 4 + wave) * 16;
  const __bf16* ar0 = sA + (lane & 15) * KK;        // pixel rows 0..15
  const __bf16* ar1 = sA + (16 + (lane & 15)) * KK; // pixel rows 16..31
  const __bf16* br  = bt + (long)(n0 + nn) * KK;
  f32x8 acc0 = {0, 0, 0, 0, 0, 0, 0, 0};
  f32x8 acc1 = {0, 0, 0, 0, 0, 0, 0, 0};
  #pragma unroll 4
  for (int kc = 0; kc < KK; kc += 32) {
    FragU fa0, fa1, fb;
    // A 16x32 bf16: lane half 0 holds K {0..7,16..23}, half 1 {8..15,24..31}
    fa0.q[0] = *reinterpret_cast<const u32x4*>(ar0 + kc + 8 * hf);
    fa0.q[1] = *reinterpret_cast<const u32x4*>(ar0 + kc + 16 + 8 * hf);
    fa1.q[0] = *reinterpret_cast<const u32x4*>(ar1 + kc + 8 * hf);
    fa1.q[1] = *reinterpret_cast<const u32x4*>(ar1 + kc + 16 + 8 * hf);
    // B 32x16: lane half 0 holds K {0..7,16..23}, half 1 {8..15,24..31}
    fb.q[0] = *reinterpret_cast<const u32x4*>(br + kc + 8 * hf);
    fb.q[1] = *reinterpret_cast<const u32x4*>(br + kc + 16 + 8 * hf);
    __builtin_prefetch(br + kc + 128, 0, 1);       // global_prefetch_b8 on weights
    acc0 = wmma_bf16(fa0.v, fb.v, acc0);
    acc1 = wmma_bf16(fa1.v, fb.v, acc1);
  }
  const float bv = bias[n0 + nn];
  #pragma unroll
  for (int j = 0; j < 8; ++j) {
    const int m = j + 8 * hf;                      // C/D layout: M = j + 8*half
    out[(pixBase + m) * COUT + n0 + nn]      = acc0[j] + bv;
    out[(pixBase + 16 + m) * COUT + n0 + nn] = acc1[j] + bv;
  }
}

// ---------- WMMA GEMM for the 1x1 quantizer conv (TDM-staged A panel) ----------
template<int KK, int NN>
__global__ __launch_bounds__(128) void k_gemm_wmma(
    const __bf16* __restrict__ A,    // (M, KK)
    const __bf16* __restrict__ bt,   // (NN, KK)
    const float*  __restrict__ bias,
    float* __restrict__ outF,        // (M, NN)
    __bf16* __restrict__ outB)       // (M, NN) bf16 copy
{
  __shared__ __align__(16) __bf16 sA[16 * KK];
  const int lane = threadIdx.x & 31;
  const int wave = threadIdx.x >> 5;
  const long pixBase = (long)blockIdx.x * 16;
#if TDM_HAVE
  if (wave == 0) {
    tdm_load_row((unsigned)(unsigned long long)sA, A + pixBase * KK, 16 * KK);
    __builtin_amdgcn_s_wait_tensorcnt(0);
  }
#else
  for (int idx = threadIdx.x; idx < (16 * KK) / 8; idx += 128)
    reinterpret_cast<u32x4*>(sA)[idx] =
        reinterpret_cast<const u32x4*>(A + pixBase * KK)[idx];
#endif
  __syncthreads();

  const int hf = lane >> 4, nn = lane & 15;
  const int n0 = (blockIdx.y * 4 + wave) * 16;
  const __bf16* ar = sA + (lane & 15) * KK;
  const __bf16* br = bt + (long)(n0 + nn) * KK;
  f32x8 acc = {0, 0, 0, 0, 0, 0, 0, 0};
  #pragma unroll
  for (int kc = 0; kc < KK; kc += 32) {
    FragU fa, fb;
    fa.q[0] = *reinterpret_cast<const u32x4*>(ar + kc + 8 * hf);
    fa.q[1] = *reinterpret_cast<const u32x4*>(ar + kc + 16 + 8 * hf);
    fb.q[0] = *reinterpret_cast<const u32x4*>(br + kc + 8 * hf);
    fb.q[1] = *reinterpret_cast<const u32x4*>(br + kc + 16 + 8 * hf);
    acc = wmma_bf16(fa.v, fb.v, acc);
  }
  const float bv = bias[n0 + nn];
  #pragma unroll
  for (int j = 0; j < 8; ++j) {
    const int m = j + 8 * hf;
    float v = acc[j] + bv;
    outF[(pixBase + m) * NN + n0 + nn] = v;
    outB[(pixBase + m) * NN + n0 + nn] = f2bf(v);
  }
}

// ---------- VQ: distances via WMMA (-2 z.c + |c|^2), argmin, gather ----------
__global__ __launch_bounds__(128) void k_vq(
    const __bf16* __restrict__ zb,     // (P3, 64) bf16
    const float*  __restrict__ zf,     // (P3, 64) fp32
    const __bf16* __restrict__ btc,    // (512, 64) bf16 codebook
    const float*  __restrict__ cbf,    // (512, 64) fp32 codebook
    const float*  __restrict__ cnorm,  // (512)
    float* __restrict__ symbols,       // NCHW (32,64,32,32)
    float* __restrict__ counts,        // (512)
    float* __restrict__ lossAccum)     // (1)
{
  __shared__ __align__(16) __bf16 sZ[16 * 64];
  __shared__ float sMinV[16][64];
  __shared__ int   sMinI[16][64];
  __shared__ int   sIdx[16];
  __shared__ float sRed[128];
  const int lane = threadIdx.x & 31;
  const int wave = threadIdx.x >> 5;
  const long pixBase = (long)blockIdx.x * 16;

#if TDM_HAVE
  if (wave == 0) {
    tdm_load_row((unsigned)(unsigned long long)sZ, zb + pixBase * 64, 16 * 64);
    __builtin_amdgcn_s_wait_tensorcnt(0);
  }
#else
  for (int idx = threadIdx.x; idx < (16 * 64) / 8; idx += 128)
    reinterpret_cast<u32x4*>(sZ)[idx] =
        reinterpret_cast<const u32x4*>(zb + pixBase * 64)[idx];
#endif
  __syncthreads();

  const int hf = lane >> 4, nn = lane & 15;
  const __bf16* ar = sZ + (lane & 15) * 64;
  float bestV[8];
  int bestI[8];
  #pragma unroll
  for (int j = 0; j < 8; ++j) { bestV[j] = 3.0e38f; bestI[j] = 0; }

  for (int nt = 0; nt < 8; ++nt) {        // each wave scans 128 codes
    const int n0 = wave * 128 + nt * 16;
    const __bf16* br = btc + (long)(n0 + nn) * 64;
    f32x8 acc = {0, 0, 0, 0, 0, 0, 0, 0};
    #pragma unroll
    for (int kc = 0; kc < 64; kc += 32) {
      FragU fa, fb;
      fa.q[0] = *reinterpret_cast<const u32x4*>(ar + kc + 8 * hf);
      fa.q[1] = *reinterpret_cast<const u32x4*>(ar + kc + 16 + 8 * hf);
      fb.q[0] = *reinterpret_cast<const u32x4*>(br + kc + 8 * hf);
      fb.q[1] = *reinterpret_cast<const u32x4*>(br + kc + 16 + 8 * hf);
      acc = wmma_bf16(fa.v, fb.v, acc);
    }
    const float cn = cnorm[n0 + nn];
    #pragma unroll
    for (int j = 0; j < 8; ++j) {
      const float d = cn - 2.0f * acc[j];  // |z|^2 constant per pixel: drop
      if (d < bestV[j]) { bestV[j] = d; bestI[j] = n0 + nn; }
    }
  }
  #pragma unroll
  for (int j = 0; j < 8; ++j) {
    const int m = j + 8 * hf;
    sMinV[m][wave * 16 + nn] = bestV[j];
    sMinI[m][wave * 16 + nn] = bestI[j];
  }
  __syncthreads();

  if (threadIdx.x < 16) {
    float bv = sMinV[threadIdx.x][0];
    int bi = sMinI[threadIdx.x][0];
    for (int c = 1; c < 64; ++c) {
      float v = sMinV[threadIdx.x][c];
      int i = sMinI[threadIdx.x][c];
      if (v < bv || (v == bv && i < bi)) { bv = v; bi = i; }
    }
    sIdx[threadIdx.x] = bi;
    atomicAdd(&counts[bi], 1.0f);
  }
  __syncthreads();

  float lsum = 0.0f;
  for (int e = threadIdx.x; e < 16 * 64; e += 128) {
    const int p = e >> 6, d = e & 63;
    const int idx = sIdx[p];
    const long P = pixBase + p;
    const float q = cbf[idx * 64 + d];
    const float zv = zf[P * 64 + d];
    const float diff = q - zv;
    lsum += diff * diff;
    const long bimg = P >> 10;
    const long rem = P & 1023;
    symbols[(bimg * 64 + d) * 1024 + rem] = q;   // straight-through fwd == codebook
  }
  sRed[threadIdx.x] = lsum;
  __syncthreads();
  for (int s = 64; s > 0; s >>= 1) {
    if (threadIdx.x < s) sRed[threadIdx.x] += sRed[threadIdx.x + s];
    __syncthreads();
  }
  if (threadIdx.x == 0) atomicAdd(lossAccum, sRed[0]);
}

// ---------- scalars: loss (1.25*MSE since e_latent==q_latent fwd) + perplexity ----------
__global__ void k_final(const float* __restrict__ counts,
                        const float* __restrict__ lossSum,
                        float* __restrict__ out2) {
  __shared__ float red[512];
  int t = threadIdx.x;
  float p = counts[t] * (1.0f / 32768.0f);
  red[t] = p * logf(p + 1e-10f);
  __syncthreads();
  for (int s = 256; s > 0; s >>= 1) {
    if (t < s) red[t] += red[t + s];
    __syncthreads();
  }
  if (t == 0) {
    out2[0] = 1.25f * lossSum[0] / (32768.0f * 64.0f);
    out2[1] = expf(-red[0]);
  }
}

// ---------- orchestration ----------
extern "C" void kernel_launch(void* const* d_in, const int* in_sizes, int n_in,
                              void* d_out, int out_size, void* d_ws, size_t ws_size,
                              hipStream_t stream) {
  (void)in_sizes; (void)n_in; (void)out_size; (void)ws_size;
  const float* x   = (const float*)d_in[0];
  const float* w1  = (const float*)d_in[1];
  const float* b1  = (const float*)d_in[2];
  const float* g1  = (const float*)d_in[3];
  const float* be1 = (const float*)d_in[4];
  const float* w2  = (const float*)d_in[5];
  const float* b2  = (const float*)d_in[6];
  const float* g2  = (const float*)d_in[7];
  const float* be2 = (const float*)d_in[8];
  const float* w3  = (const float*)d_in[9];
  const float* b3  = (const float*)d_in[10];
  const float* g3  = (const float*)d_in[11];
  const float* be3 = (const float*)d_in[12];
  const float* wq  = (const float*)d_in[13];
  const float* bq  = (const float*)d_in[14];
  const float* cb  = (const float*)d_in[15];

  char* ws = (char*)d_ws;
  float*  raw   = (float*)(ws + OFF_RAW);
  __bf16* act1  = (__bf16*)(ws + OFF_ACT1);   // also h3 later
  __bf16* act2  = (__bf16*)(ws + OFF_ACT2);
  float*  zfp   = (float*)(ws + OFF_ZF);
  __bf16* zbf   = (__bf16*)(ws + OFF_ZB);
  __bf16* bt2   = (__bf16*)(ws + OFF_BT2);
  __bf16* bt3   = (__bf16*)(ws + OFF_BT3);
  __bf16* btq   = (__bf16*)(ws + OFF_BTQ);
  __bf16* btc   = (__bf16*)(ws + OFF_BTC);
  float*  cnorm = (float*)(ws + OFF_CNORM);
  float*  stats = (float*)(ws + OFF_STATS);
  float*  outF  = (float*)d_out;

  k_zero<<<(S_TOTAL + 255) / 256, 256, 0, stream>>>(stats, S_TOTAL);
  k_wconv<<<(128 * 64 * 16 + 255) / 256, 256, 0, stream>>>(w2, bt2, 64, 128);
  k_wconv<<<(256 * 128 * 16 + 255) / 256, 256, 0, stream>>>(w3, bt3, 128, 256);
  k_cast<<<(64 * 256 + 255) / 256, 256, 0, stream>>>(wq, btq, 64 * 256);
  k_cast<<<(512 * 64 + 255) / 256, 256, 0, stream>>>(cb, btc, 512 * 64);
  k_cnorm<<<2, 256, 0, stream>>>(cb, cnorm);

  // layer 1: direct conv -> BN stats -> BN+ReLU -> bf16 NHWC
  k_conv1<<<(int)(P1 * 64 / 256), 256, 0, stream>>>(x, w1, b1, raw);
  k_bnstats<<<512, 256, 0, stream>>>(raw, stats + S_SUM1, stats + S_SQ1, P1 * 64, 63);
  k_bnfin<<<1, 256, 0, stream>>>(stats + S_SUM1, stats + S_SQ1, g1, be1,
                                 stats + S_SC1, stats + S_SH1, 1.0f / (float)P1, 64);
  k_act<<<4096, 256, 0, stream>>>(raw, stats + S_SC1, stats + S_SH1, act1, P1 * 64, 63);

  // layer 2: WMMA conv (M=32 tiles)
  k_conv4x4s2_wmma<64, 128, 64, 64><<<dim3((unsigned)(P2 / 32), 2), 128, 0, stream>>>(
      act1, bt2, b2, raw);
  k_bnstats<<<512, 256, 0, stream>>>(raw, stats + S_SUM2, stats + S_SQ2, P2 * 128, 127);
  k_bnfin<<<1, 256, 0, stream>>>(stats + S_SUM2, stats + S_SQ2, g2, be2,
                                 stats + S_SC2, stats + S_SH2, 1.0f / (float)P2, 128);
  k_act<<<4096, 256, 0, stream>>>(raw, stats + S_SC2, stats + S_SH2, act2, P2 * 128, 127);

  // layer 3: WMMA conv (h3 reuses the act1 region; h1 is dead)
  k_conv4x4s2_wmma<128, 256, 32, 32><<<dim3((unsigned)(P3 / 32), 4), 128, 0, stream>>>(
      act2, bt3, b3, raw);
  k_bnstats<<<512, 256, 0, stream>>>(raw, stats + S_SUM3, stats + S_SQ3, P3 * 256, 255);
  k_bnfin<<<1, 256, 0, stream>>>(stats + S_SUM3, stats + S_SQ3, g3, be3,
                                 stats + S_SC3, stats + S_SH3, 1.0f / (float)P3, 256);
  k_act<<<4096, 256, 0, stream>>>(raw, stats + S_SC3, stats + S_SH3, act1, P3 * 256, 255);

  // quantizer 1x1 conv (GEMM M=32768 K=256 N=64) -> z fp32 + z bf16
  k_gemm_wmma<256, 64><<<dim3((unsigned)(P3 / 16), 1), 128, 0, stream>>>(
      act1, btq, bq, zfp, zbf);

  // vector quantization + symbols + counts + loss accumulation
  k_vq<<<(unsigned)(P3 / 16), 128, 0, stream>>>(zbf, zfp, btc, cb, cnorm, outF,
                                                stats + S_CNT, stats + S_LOSS);

  // scalars
  k_final<<<1, 512, 0, stream>>>(stats + S_CNT, stats + S_LOSS, outF + 2097152);
}
